// PiKVMoE_652835029299
// MI455X (gfx1250) — compile-verified, matrix operand order checked by reading, not run on hardware
//
#include <hip/hip_runtime.h>

// ---------------- problem constants (match reference) ----------------
#define HID       1024
#define VOC       32000
#define NE        8
#define RK        4
#define KVC       2048
#define CACHE_DEC 128
#define NTOK      2048          // B*S = 2*1024
#define LORA_SCALE 0.25f        // ALPHA / RANK
#define BSTRIDE   40            // LDS tile row stride in halfwords (80 B, 16B-aligned)

typedef __attribute__((ext_vector_type(16))) __bf16 v16bf;
typedef __attribute__((ext_vector_type(8)))  __bf16 v8bf;
typedef __attribute__((ext_vector_type(8)))  float  v8f;
typedef __attribute__((ext_vector_type(4)))  unsigned int v4u;
typedef __attribute__((ext_vector_type(8)))  int v8i;
typedef __attribute__((ext_vector_type(4)))  int v4i;

// ------------------------------------------------------------------
// v_wmma_f32_16x16x32_bf16 fragment layouts (wave32):
// A (16x32 MxK): lane%16 = M row, lane/16 = K-half (hi).
//   elem j: K = 8*hi + j (j<8) ; K = 8*hi + 16 + (j-8) (j>=8)
// B (32x16 KxN): lane%16 = N col, lane/16 = K-half; same K mapping.
// C/D (16x16 f32): VGPR r, lane L: M = r + 8*(L/16), N = L%16.
// ------------------------------------------------------------------
__device__ inline v16bf load_frag_rowk(const __bf16* base /* row + 8*hi halfs */) {
  v8bf lo = *(const v8bf*)(base);        // K = 8*hi + 0..7
  v8bf hh = *(const v8bf*)(base + 16);   // K = 8*hi + 16..23
  v16bf f;
#pragma unroll
  for (int j = 0; j < 8; ++j) { f[j] = lo[j]; f[j + 8] = hh[j]; }
  return f;
}

// pack two f32 -> one dword of two bf16 (k, k+1) and store to LDS
__device__ inline void store_pk_bf16(__bf16* dst, float f0, float f1) {
  unsigned int u = ((unsigned int)__builtin_bit_cast(unsigned short, (__bf16)f1) << 16) |
                   (unsigned int)__builtin_bit_cast(unsigned short, (__bf16)f0);
  *(unsigned int*)dst = u;
}

// ------------------------------------------------------------------
// TDM: async DMA of a [128 x 32] bf16 tile (row-major, row pitch HID)
// into LDS with 4-dword padding after each 16-dword row -> 80B LDS pitch.
// D# built per CDNA5 ISA ch.8 (group0/1; 2-D tile so groups 2/3 zero).
// This toolchain exposes the 6-arg builtin:
//   (uint32x4 g0, int32x8 g1, int32x4, int32x4, int32x8, i32 cpol)
// ------------------------------------------------------------------
__device__ inline void tdm_load_a_tile(const void* gaddr, unsigned ldsByteOff) {
  unsigned long long ga = (unsigned long long)(size_t)gaddr;
  v4u g0;
  g0[0] = 1u;                                            // count=1, user D#
  g0[1] = ldsByteOff;                                    // lds_addr (bytes)
  g0[2] = (unsigned)(ga & 0xFFFFFFFFu);                  // global_addr[31:0]
  g0[3] = (unsigned)((ga >> 32) & 0x01FFFFFFu)           // global_addr[56:32]
          | 0x80000000u;                                 // type=2 ("image")
  v8i g1;
  g1[0] = 0x06D10000;        // data_size=2B | pad_en | pad_interval=16dw | pad_amount=4dw
  g1[1] = (int)(HID << 16);  // tensor_dim0 = 1024 (bits 79:48, low 16 here)
  g1[2] = (int)((NTOK & 0xFFFF) << 16);  // tensor_dim1 = 2048 (low 16)
  g1[3] = (int)(32 << 16);   // tensor_dim1 hi=0 | tile_dim0 = 32
  g1[4] = 128;               // tile_dim1 = 128, tile_dim2 = 0
  g1[5] = HID;               // tensor_dim0_stride = 1024 (low 32)
  g1[6] = 0;
  g1[7] = 0;                 // tensor_dim1_stride unused (2-D tile)
  v4i z4 = {0, 0, 0, 0};
  v8i z8 = {0, 0, 0, 0, 0, 0, 0, 0};
  __builtin_amdgcn_tensor_load_to_lds(g0, g1, z4, z4, z8, 0);
}

// ------------------------------------------------------------------
// Kernel 1: pyramidal cache vectors.  grid=8 (one block / expert), 256 thr.
// ------------------------------------------------------------------
__global__ __launch_bounds__(256) void k_cache(const float* __restrict__ cv,
                                               const float* __restrict__ ci,
                                               const float* __restrict__ cla,
                                               const float* __restrict__ clb,
                                               float* __restrict__ out) {
  __shared__ float         s_imp[KVC];
  __shared__ unsigned char s_keep[KVC];
  __shared__ float         s_mean[HID];
  __shared__ float         s_red[256 * RK];
  __shared__ float         s_t[RK];

  const int e   = blockIdx.x;
  const int tid = threadIdx.x;
  const int n   = KVC - CACHE_DEC * e;
  const int k   = n / 2;

  for (int i = tid; i < n; i += 256) s_imp[i] = ci[(size_t)e * KVC + i];
  __syncthreads();

  for (int i = tid; i < n; i += 256) {
    float v = s_imp[i];
    int rank = 0;
    for (int j = 0; j < n; ++j) {
      float u = s_imp[j];
      rank += (u > v) || (u == v && j < i);
    }
    s_keep[i] = (rank < k) ? 1 : 0;
  }
  __syncthreads();

  float acc[4] = {0.f, 0.f, 0.f, 0.f};
  for (int j = 0; j < n; ++j) {
    if (s_keep[j]) {
      const float* row = cv + ((size_t)e * KVC + j) * HID;
#pragma unroll
      for (int q = 0; q < 4; ++q) acc[q] += row[tid + 256 * q];
    }
  }
  const float inv = 1.0f / (float)k;
#pragma unroll
  for (int q = 0; q < 4; ++q) s_mean[tid + 256 * q] = acc[q] * inv;
  __syncthreads();

  float p[RK] = {0.f, 0.f, 0.f, 0.f};
#pragma unroll
  for (int q = 0; q < 4; ++q) {
    int h = tid + 256 * q;
    const float* la = cla + ((size_t)e * HID + h) * RK;
    float mv = s_mean[h];
#pragma unroll
    for (int r = 0; r < RK; ++r) p[r] += mv * la[r];
  }
#pragma unroll
  for (int r = 0; r < RK; ++r) s_red[tid * RK + r] = p[r];
  __syncthreads();
  for (int off = 128; off > 0; off >>= 1) {
    if (tid < off)
#pragma unroll
      for (int r = 0; r < RK; ++r) s_red[tid * RK + r] += s_red[(tid + off) * RK + r];
    __syncthreads();
  }
  if (tid < RK) s_t[tid] = s_red[tid];
  __syncthreads();

#pragma unroll
  for (int q = 0; q < 4; ++q) {
    int h = tid + 256 * q;
    float lo = 0.f;
#pragma unroll
    for (int r = 0; r < RK; ++r) lo += s_t[r] * clb[(size_t)e * RK * HID + (size_t)r * HID + h];
    out[(size_t)e * HID + h] = s_mean[h] + LORA_SCALE * lo;
  }
}

// ------------------------------------------------------------------
// Kernel 2: per-token embed + router + u = h@la.  grid=2048, 256 thr.
// ------------------------------------------------------------------
__global__ __launch_bounds__(256) void k_token(const int*   __restrict__ x,
                                               const float* __restrict__ emb,
                                               const float* __restrict__ gw,
                                               const float* __restrict__ gb,
                                               const float* __restrict__ la,
                                               __bf16* __restrict__ hB,
                                               float*  __restrict__ route,
                                               float*  __restrict__ uArr) {
  __shared__ float s_red[256 * NE];
  const int t   = blockIdx.x;
  const int tid = threadIdx.x;
  const int tok = x[t];
  const float* erow = emb + (size_t)tok * HID;

  float h4[4];
#pragma unroll
  for (int q = 0; q < 4; ++q) {
    h4[q] = erow[tid + 256 * q];
    hB[(size_t)t * HID + tid + 256 * q] = (__bf16)h4[q];
  }

  float g[NE];
#pragma unroll
  for (int e = 0; e < NE; ++e) g[e] = 0.f;
#pragma unroll
  for (int q = 0; q < 4; ++q) {
    const float* gr = gw + (size_t)(tid + 256 * q) * NE;
#pragma unroll
    for (int e = 0; e < NE; ++e) g[e] += h4[q] * gr[e];
  }
  float up[NE * RK];
#pragma unroll
  for (int i = 0; i < NE * RK; ++i) up[i] = 0.f;
#pragma unroll
  for (int q = 0; q < 4; ++q) {
    int h = tid + 256 * q;
#pragma unroll
    for (int e = 0; e < NE; ++e) {
      const float* lr = la + ((size_t)e * HID + h) * RK;
#pragma unroll
      for (int r = 0; r < RK; ++r) up[e * RK + r] += h4[q] * lr[r];
    }
  }

#pragma unroll
  for (int e = 0; e < NE; ++e) s_red[tid * NE + e] = g[e];
  __syncthreads();
  for (int off = 128; off > 0; off >>= 1) {
    if (tid < off)
#pragma unroll
      for (int e = 0; e < NE; ++e) s_red[tid * NE + e] += s_red[(tid + off) * NE + e];
    __syncthreads();
  }
  if (tid == 0) {
    float ge[NE], pr[NE];
    float m = -1e30f;
#pragma unroll
    for (int e = 0; e < NE; ++e) { ge[e] = s_red[e] + gb[e]; m = fmaxf(m, ge[e]); }
    float s = 0.f;
#pragma unroll
    for (int e = 0; e < NE; ++e) { pr[e] = __expf(ge[e] - m); s += pr[e]; }
#pragma unroll
    for (int e = 0; e < NE; ++e) pr[e] /= s;
    int i0 = 0;
#pragma unroll
    for (int e = 1; e < NE; ++e) if (pr[e] > pr[i0]) i0 = e;
    int i1 = (i0 == 0) ? 1 : 0;
#pragma unroll
    for (int e = 0; e < NE; ++e) if (e != i0 && pr[e] > pr[i1]) i1 = e;
    float s2 = pr[i0] + pr[i1];
#pragma unroll
    for (int e = 0; e < NE; ++e) route[(size_t)t * NE + e] = 0.f;
    route[(size_t)t * NE + i0] = pr[i0] / s2;
    route[(size_t)t * NE + i1] = pr[i1] / s2;
  }
  __syncthreads();

  for (int c = 0; c < 4; ++c) {
#pragma unroll
    for (int j = 0; j < 8; ++j) s_red[tid * 8 + j] = up[c * 8 + j];
    __syncthreads();
    for (int off = 128; off > 0; off >>= 1) {
      if (tid < off)
#pragma unroll
        for (int j = 0; j < 8; ++j) s_red[tid * 8 + j] += s_red[(tid + off) * 8 + j];
      __syncthreads();
    }
    if (tid == 0)
#pragma unroll
      for (int j = 0; j < 8; ++j) uArr[(size_t)t * (NE * RK) + c * 8 + j] = s_red[j];
    __syncthreads();
  }
}

// ------------------------------------------------------------------
// Kernel 3: all-expert WMMA GEMM + routed combine -> X (bf16).
// grid (16,16): block = 128 tokens x 64 features; wave = 16x64 strip.
// ------------------------------------------------------------------
__global__ __launch_bounds__(256) void k_expert(const __bf16* __restrict__ hB,
                                                const float* __restrict__ ew,
                                                const float* __restrict__ eb,
                                                const float* __restrict__ lb,
                                                const float* __restrict__ route,
                                                const float* __restrict__ uArr,
                                                const float* __restrict__ cvec,
                                                __bf16* __restrict__ xB) {
  __shared__ __bf16 sB[64 * BSTRIDE];    // 5 KB
  const int tid  = threadIdx.x;
  const int wave = tid >> 5, lane = tid & 31;
  const int hi = lane >> 4, l16 = lane & 15;
  const int mBase = blockIdx.x * 128 + wave * 16;
  const int nBase = blockIdx.y * 64;
  const int kk = tid >> 4;           // 0..15 -> k = 2*kk, 2*kk+1
  const int n4 = tid & 15;           // 0..15 -> n = n4*4 .. +3

  v8f xacc[4];
#pragma unroll
  for (int nt = 0; nt < 4; ++nt) xacc[nt] = {};

  const __bf16* arow = hB + (size_t)(mBase + l16) * HID + 8 * hi;

  for (int e = 0; e < NE; ++e) {
    v8f acc[4];
#pragma unroll
    for (int nt = 0; nt < 4; ++nt) acc[nt] = {};
    const float* wbase = ew + (size_t)e * HID * HID;

    for (int k0 = 0; k0 < HID; k0 += 32) {
      const float4 f0 = *(const float4*)(wbase + (size_t)(k0 + 2 * kk) * HID + nBase + n4 * 4);
      const float4 f1 = *(const float4*)(wbase + (size_t)(k0 + 2 * kk + 1) * HID + nBase + n4 * 4);
      store_pk_bf16(&sB[(n4 * 4 + 0) * BSTRIDE + 2 * kk], f0.x, f1.x);
      store_pk_bf16(&sB[(n4 * 4 + 1) * BSTRIDE + 2 * kk], f0.y, f1.y);
      store_pk_bf16(&sB[(n4 * 4 + 2) * BSTRIDE + 2 * kk], f0.z, f1.z);
      store_pk_bf16(&sB[(n4 * 4 + 3) * BSTRIDE + 2 * kk], f0.w, f1.w);
      __syncthreads();

      v16bf a = load_frag_rowk(arow + k0);
#pragma unroll
      for (int nt = 0; nt < 4; ++nt) {
        v16bf b = load_frag_rowk(sB + (size_t)(nt * 16 + l16) * BSTRIDE + 8 * hi);
        acc[nt] = __builtin_amdgcn_wmma_f32_16x16x32_bf16(
            false, a, false, b, (short)0, acc[nt], false, false);
      }
      __syncthreads();
    }

    float wrt[8], uu[8][RK];
#pragma unroll
    for (int r = 0; r < 8; ++r) {
      int tcur = mBase + r + 8 * hi;
      wrt[r] = route[(size_t)tcur * NE + e];
#pragma unroll
      for (int rr = 0; rr < RK; ++rr)
        uu[r][rr] = uArr[(size_t)tcur * (NE * RK) + e * RK + rr];
    }
#pragma unroll
    for (int nt = 0; nt < 4; ++nt) {
      int n = nBase + nt * 16 + l16;
      float base = eb[(size_t)e * HID + n] + cvec[(size_t)e * HID + n];
      float lb0 = lb[(size_t)e * RK * HID + 0 * HID + n];
      float lb1 = lb[(size_t)e * RK * HID + 1 * HID + n];
      float lb2 = lb[(size_t)e * RK * HID + 2 * HID + n];
      float lb3 = lb[(size_t)e * RK * HID + 3 * HID + n];
#pragma unroll
      for (int r = 0; r < 8; ++r) {
        float val = acc[nt][r] + base +
                    LORA_SCALE * (uu[r][0] * lb0 + uu[r][1] * lb1 +
                                  uu[r][2] * lb2 + uu[r][3] * lb3);
        xacc[nt][r] += wrt[r] * val;
      }
    }
  }

#pragma unroll
  for (int nt = 0; nt < 4; ++nt) {
    int n = nBase + nt * 16 + l16;
#pragma unroll
    for (int r = 0; r < 8; ++r) {
      int tcur = mBase + r + 8 * hi;
      xB[(size_t)tcur * HID + n] = (__bf16)xacc[nt][r];
    }
  }
}

// ------------------------------------------------------------------
// Kernel 4: vocab WMMA GEMM: X(bf16) @ vocab_w + vocab_b -> logits f32.
// grid (16,250): block = 128 tokens x 128 vocab; wave = 16x128 strip.
// A tile (128x32 bf16) DMA'd by the Tensor Data Mover into LDS,
// double-buffered (TENSORcnt overlap). B tile staged via cvt+LDS.
// ------------------------------------------------------------------
__global__ __launch_bounds__(256) void k_vocab(const __bf16* __restrict__ xB,
                                               const float* __restrict__ vw,
                                               const float* __restrict__ vb,
                                               float* __restrict__ out) {
  __shared__ __bf16 sA[2 * 128 * BSTRIDE];   // 2 x 10 KB (TDM pads to 80B pitch)
  __shared__ __bf16 sB[128 * BSTRIDE];       // 10 KB
  const int tid  = threadIdx.x;
  const int wave = tid >> 5, lane = tid & 31;
  const int hi = lane >> 4, l16 = lane & 15;
  const int mBase = blockIdx.x * 128;
  const int nBase = blockIdx.y * 128;
  const int kk  = tid >> 4;           // 0..15
  const int n4a = tid & 15;           // n-quad 0..15
  const int n4b = (tid & 15) + 16;    // n-quad 16..31

  const unsigned aOff0 = (unsigned)(size_t)(&sA[0]);
  const __bf16* aTile = xB + (size_t)mBase * HID;   // block's A rows

  v8f acc[8];
#pragma unroll
  for (int nt = 0; nt < 8; ++nt) acc[nt] = {};

  // prologue: DMA first A K-slab into buffer 0
  if (wave == 0) tdm_load_a_tile(aTile, aOff0);

  for (int ks = 0; ks < HID / 32; ++ks) {
    const int k0 = ks * 32;
    const int cur = ks & 1;

    // ---- cooperative B fill: coalesced b128 loads, packed bf16 LDS stores
#pragma unroll
    for (int u = 0; u < 2; ++u) {
      const int n4 = (u == 0) ? n4a : n4b;
      const float* r0 = vw + (size_t)(k0 + 2 * kk) * VOC + nBase + n4 * 4;
      const float4 f0 = *(const float4*)(r0);
      const float4 f1 = *(const float4*)(r0 + VOC);
      if (k0 + 32 < HID) __builtin_prefetch(r0 + (size_t)32 * VOC, 0, 1);
      store_pk_bf16(&sB[(n4 * 4 + 0) * BSTRIDE + 2 * kk], f0.x, f1.x);
      store_pk_bf16(&sB[(n4 * 4 + 1) * BSTRIDE + 2 * kk], f0.y, f1.y);
      store_pk_bf16(&sB[(n4 * 4 + 2) * BSTRIDE + 2 * kk], f0.z, f1.z);
      store_pk_bf16(&sB[(n4 * 4 + 3) * BSTRIDE + 2 * kk], f0.w, f1.w);
    }

    // ---- wave 0: kick next A DMA, wait for current one (in-order per wave)
    if (wave == 0) {
      if (ks + 1 < HID / 32) {
        tdm_load_a_tile(aTile + (k0 + 32),
                        aOff0 + (unsigned)((cur ^ 1) * 128 * BSTRIDE * 2));
        __builtin_amdgcn_s_wait_tensorcnt(1);
      } else {
        __builtin_amdgcn_s_wait_tensorcnt(0);
      }
    }
    __syncthreads();   // publish B stores + A DMA to the workgroup

    const __bf16* aBuf = sA + (size_t)cur * 128 * BSTRIDE;
    v16bf a = load_frag_rowk(aBuf + (size_t)(wave * 16 + l16) * BSTRIDE + 8 * hi);
#pragma unroll
    for (int nt = 0; nt < 8; ++nt) {
      v16bf b = load_frag_rowk(sB + (size_t)(nt * 16 + l16) * BSTRIDE + 8 * hi);
      acc[nt] = __builtin_amdgcn_wmma_f32_16x16x32_bf16(
          false, a, false, b, (short)0, acc[nt], false, false);
    }
    __syncthreads();   // tiles consumed; safe to refill
  }

#pragma unroll
  for (int nt = 0; nt < 8; ++nt) {
    int n = nBase + nt * 16 + l16;
    float bias = vb[n];
#pragma unroll
    for (int r = 0; r < 8; ++r) {
      int tcur = mBase + wave * 16 + r + 8 * hi;
      out[(size_t)tcur * VOC + n] = acc[nt][r] + bias;
    }
  }
}

// ------------------------------------------------------------------
extern "C" void kernel_launch(void* const* d_in, const int* in_sizes, int n_in,
                              void* d_out, int out_size, void* d_ws, size_t ws_size,
                              hipStream_t stream) {
  const int*   x     = (const int*)d_in[0];
  const float* emb   = (const float*)d_in[1];
  const float* gw    = (const float*)d_in[2];
  const float* gb    = (const float*)d_in[3];
  const float* ew    = (const float*)d_in[4];
  const float* eb    = (const float*)d_in[5];
  const float* ela   = (const float*)d_in[6];
  const float* elb   = (const float*)d_in[7];
  const float* cval  = (const float*)d_in[8];
  const float* cimp  = (const float*)d_in[9];
  const float* cla   = (const float*)d_in[10];
  const float* clb   = (const float*)d_in[11];
  const float* vw    = (const float*)d_in[12];
  const float* vb    = (const float*)d_in[13];
  float* out = (float*)d_out;

  char* wsb = (char*)d_ws;
  __bf16* ws_h     = (__bf16*)(wsb);                                   // 4 MB
  __bf16* ws_xb    = (__bf16*)(wsb + (size_t)4 * 1024 * 1024);         // 4 MB
  float*  ws_route = (float*)(wsb + (size_t)8 * 1024 * 1024);          // 64 KB
  float*  ws_u     = (float*)(wsb + (size_t)8 * 1024 * 1024 + 65536);  // 256 KB
  float*  ws_cvec  = (float*)(wsb + (size_t)8 * 1024 * 1024 + 65536 + 262144); // 32 KB

  k_cache<<<NE, 256, 0, stream>>>(cval, cimp, cla, clb, ws_cvec);
  k_token<<<NTOK, 256, 0, stream>>>(x, emb, gw, gb, ela, ws_h, ws_route, ws_u);
  k_expert<<<dim3(NTOK / 128, HID / 64), 256, 0, stream>>>(
      ws_h, ew, eb, elb, ws_route, ws_u, ws_cvec, ws_xb);
  k_vocab<<<dim3(NTOK / 128, VOC / 128), 256, 0, stream>>>(ws_xb, vw, vb, out);
}
